// TransformerBlock_11974368821598
// MI455X (gfx1250) — matrix operand order repaired and checked by reading, not compiled, not run on hardware
//
#include <hip/hip_runtime.h>
#include <hip/hip_bf16.h>

// ---------------------------------------------------------------------------
// Transformer block for MI455X (gfx1250): bf16 WMMA GEMMs (TDM-staged A tiles,
// double-buffered LDS) + causal flash attention.
// B=2, T=2048, D=1024, H=16, HS=64, DFF=4096.
// ---------------------------------------------------------------------------

typedef __attribute__((ext_vector_type(8)))  float    v8f;
typedef __attribute__((ext_vector_type(16))) __bf16   v16bf;
typedef __attribute__((ext_vector_type(8)))  __bf16   v8bf;
typedef __attribute__((ext_vector_type(4)))  unsigned v4u;
typedef __attribute__((ext_vector_type(8)))  unsigned v8u;

#define BB  2
#define TT  2048
#define DD  1024
#define HH  16
#define HSZ 64
#define DFF 4096
#define MM  (BB * TT)

// LDS A-tile: 64 rows x 64 K elements, padded row pitch 72 (TDM pad: every
// 32 DWORDs insert 4 DWORDs) -> 16-byte aligned rows, bank-conflict friendly.
#define APITCH 72

__device__ __forceinline__ v8f wmma_bf16(v16bf a, v16bf b, v8f c) {
  return __builtin_amdgcn_wmma_f32_16x16x32_bf16(
      /*neg_a=*/false, a, /*neg_b=*/false, b,
      /*c_mod=*/(short)0, c, /*reuse_a=*/false, /*reuse_b=*/false);
}

// A-matrix fragment: lane (l&15) holds row m; lane-half h16 selects the
// K range: lanes 0-15 -> K {0..15}, lanes 16-31 -> K {16..31}.
__device__ __forceinline__ v16bf pack_a(const __bf16* p, int h16) {
  v8bf lo = *(const v8bf*)(p + h16 * 16);
  v8bf hi = *(const v8bf*)(p + h16 * 16 + 8);
  v16bf a;
#pragma unroll
  for (int i = 0; i < 8; ++i) { a[i] = lo[i]; a[i + 8] = hi[i]; }
  return a;
}

__device__ __forceinline__ float grp16_max(float v) {
  v = fmaxf(v, __shfl_xor(v, 1, 32));
  v = fmaxf(v, __shfl_xor(v, 2, 32));
  v = fmaxf(v, __shfl_xor(v, 4, 32));
  v = fmaxf(v, __shfl_xor(v, 8, 32));
  return v;
}
__device__ __forceinline__ float grp16_sum(float v) {
  v += __shfl_xor(v, 1, 32);
  v += __shfl_xor(v, 2, 32);
  v += __shfl_xor(v, 4, 32);
  v += __shfl_xor(v, 8, 32);
  return v;
}

// ---------------------------------------------------------------------------
// TDM: async 2D tile load Global -> LDS (D# per CDNA5 ISA).
// 64 x 64 bf16 tile; tensor row stride = strideElems; LDS pad gives
// APITCH-element row pitch.  Issued once per wave (EXEC ignored by TDM).
// ---------------------------------------------------------------------------
__device__ __forceinline__ void tdm_load_a_tile(unsigned ldsAddr, const __bf16* g,
                                                unsigned strideElems) {
  unsigned long long ga = (unsigned long long)(void*)g;
  v4u g0;
  g0[0] = 1u;                                   // count=1 (valid user descriptor)
  g0[1] = ldsAddr;                              // LDS byte address
  g0[2] = (unsigned)(ga & 0xffffffffu);         // global_addr[31:0]
  g0[3] = (unsigned)((ga >> 32) & 0x1ffffffu)   // global_addr[56:32]
        | (2u << 30);                           // type=2 ("image")
  v8u g1;
  g1[0] = (1u << 16)        // data_size=1 -> 2 bytes
        | (1u << 20)        // pad_enable
        | (4u << 22)        // pad_interval: code 4 -> every 32 DWORDs (64 bf16)
        | (3u << 25);       // pad_amount:   code 3 -> 4 DWORDs (8 bf16)
  g1[1] = (strideElems & 0xffffu) << 16;        // tensor_dim0[15:0]
  g1[2] = (strideElems >> 16);                  // tensor_dim0[31:16]; tensor_dim1[15:0]=0
  g1[3] = 16u                                   // tensor_dim1[31:16] -> dim1 = 1<<20 rows
        | (64u << 16);                          // tile_dim0 = 64 elements
  g1[4] = 64u;                                  // tile_dim1 = 64 rows; tile_dim2 = 0
  g1[5] = strideElems;                          // tensor_dim0_stride[31:0]
  g1[6] = 0u;                                   // stride hi = 0; dim1_stride lo = 0
  g1[7] = 0u;
  asm volatile("tensor_load_to_lds %0, %1, null, null"
               :: "s"(g0), "s"(g1)
               : "memory");
}

// ---------------------------------------------------------------------------
// Weight cast + transpose: in[K,N] f32 -> out[N,K] bf16  (B-fragment friendly)
// ---------------------------------------------------------------------------
__global__ __launch_bounds__(256) void convT_kernel(
    const float* __restrict__ in, __bf16* __restrict__ out, int K, int N) {
  size_t idx = (size_t)blockIdx.x * 256 + threadIdx.x;
  if (idx >= (size_t)K * N) return;
  int n = (int)(idx % N);
  int k = (int)(idx / N);
  out[(size_t)n * K + k] = (__bf16)in[idx];
}

// ---------------------------------------------------------------------------
// V transpose: v[B,T,H,HS] bf16 -> vt[B,H,HS,T] bf16
// ---------------------------------------------------------------------------
__global__ __launch_bounds__(256) void transV_kernel(
    const __bf16* __restrict__ v, __bf16* __restrict__ vtOut) {
  size_t idx = (size_t)blockIdx.x * 256 + threadIdx.x;
  if (idx >= (size_t)MM * DD) return;
  int d = (int)(idx % DD);
  int m = (int)(idx / DD);
  int b = m / TT, t = m % TT;
  int head = d / HSZ, hs = d % HSZ;
  vtOut[((size_t)(b * HH + head) * HSZ + hs) * TT + t] = v[idx];
}

// ---------------------------------------------------------------------------
// LayerNorm: x[M,D] f32 -> out[M,D] bf16 (one block per row)
// ---------------------------------------------------------------------------
__global__ __launch_bounds__(256) void ln_kernel(
    const float* __restrict__ x, const float* __restrict__ g,
    const float* __restrict__ bta, __bf16* __restrict__ out) {
  const int row = blockIdx.x, tid = threadIdx.x;
  const float* xr = x + (size_t)row * DD;
  float v[4], s = 0.f, s2 = 0.f;
#pragma unroll
  for (int j = 0; j < 4; ++j) {
    v[j] = xr[tid + 256 * j];
    s += v[j];
    s2 += v[j] * v[j];
  }
  __shared__ float rs[256], rq[256];
  rs[tid] = s; rq[tid] = s2;
  __syncthreads();
  for (int off = 128; off > 0; off >>= 1) {
    if (tid < off) { rs[tid] += rs[tid + off]; rq[tid] += rq[tid + off]; }
    __syncthreads();
  }
  float mu  = rs[0] * (1.f / DD);
  float var = rq[0] * (1.f / DD) - mu * mu;
  float inv = rsqrtf(var + 1e-5f);
  __bf16* orow = out + (size_t)row * DD;
#pragma unroll
  for (int j = 0; j < 4; ++j) {
    int idx = tid + 256 * j;
    orow[idx] = (__bf16)((v[j] - mu) * inv * g[idx] + bta[idx]);
  }
}

// ---------------------------------------------------------------------------
// bf16 WMMA GEMM: C[M,N] = A[M,K] * Wt[N,K]^T (+bias, relu, residual).
// Block = 256 threads = 8 waves (2 M-waves x 4 N-waves), block tile 64 x 256,
// wave tile 32 x 64.  A block-tile (64x64) streamed by TDM into
// double-buffered LDS; B fragments direct from global (L2-resident weights).
// ---------------------------------------------------------------------------
__global__ __launch_bounds__(256) void gemm_bf16_kernel(
    const __bf16* __restrict__ A, const __bf16* __restrict__ Wt,
    const float* __restrict__ bias, const float* __restrict__ res,
    float* __restrict__ outF, __bf16* __restrict__ outB,
    int M, int N, int K, int relu) {
  __shared__ __bf16 atile[2][64 * APITCH];   // 2 x 9KB

  const int lane = threadIdx.x & 31;
  const int wid  = threadIdx.x >> 5;
  const int h16  = lane >> 4;
  const int ln   = lane & 15;
  const int wm   = wid >> 2;          // 0..1  (M wave)
  const int wn   = wid & 3;           // 0..3  (N wave)
  const int mBase = blockIdx.y * 64;
  const int nBase = blockIdx.x * 256 + wn * 64;
  const int mW    = mBase + wm * 32;

  const __bf16* bRow[4];
#pragma unroll
  for (int j = 0; j < 4; ++j)
    bRow[j] = Wt + (size_t)(nBase + j * 16 + ln) * K;

  v8f acc[2][4] = {};

  const int nChunks = K / 64;
  // Prologue: kick off chunk 0.
  if (wid == 0) {
    tdm_load_a_tile((unsigned)(unsigned long long)(void*)&atile[0][0],
                    A + (size_t)mBase * K, (unsigned)K);
  }

  for (int c = 0; c < nChunks; ++c) {
    const int cur = c & 1;
    if (wid == 0) __builtin_amdgcn_s_wait_tensorcnt(0);
    __syncthreads();                   // chunk c resident in atile[cur]
    if (wid == 0 && (c + 1) < nChunks) {
      // Overlap: DMA next chunk into the other buffer while we compute.
      tdm_load_a_tile((unsigned)(unsigned long long)(void*)&atile[cur ^ 1][0],
                      A + (size_t)mBase * K + (size_t)(c + 1) * 64, (unsigned)K);
    }
    if ((c + 1) < nChunks) {
      // L2 prefetch of the B streams one chunk ahead.
#pragma unroll
      for (int j = 0; j < 4; ++j)
        __builtin_prefetch(bRow[j] + (size_t)(c + 1) * 64, 0, 1);
    }

    const __bf16* at = &atile[cur][0];
#pragma unroll
    for (int kk = 0; kk < 64; kk += 32) {
      const int kg = c * 64 + kk;
      // Issue the whole k-step's loads first (one clause), then 8 WMMAs.
      v16bf bfr[4];
#pragma unroll
      for (int j = 0; j < 4; ++j)
        bfr[j] = *(const v16bf*)(bRow[j] + kg + h16 * 16);
      v16bf a0 = pack_a(at + (wm * 32 + ln) * APITCH + kk, h16);
      v16bf a1 = pack_a(at + (wm * 32 + 16 + ln) * APITCH + kk, h16);
#pragma unroll
      for (int j = 0; j < 4; ++j) {
        acc[0][j] = wmma_bf16(a0, bfr[j], acc[0][j]);
        acc[1][j] = wmma_bf16(a1, bfr[j], acc[1][j]);
      }
    }
    __syncthreads();                   // done reading atile[cur] before reuse
  }

#pragma unroll
  for (int i2 = 0; i2 < 2; ++i2) {
#pragma unroll
    for (int j = 0; j < 4; ++j) {
#pragma unroll
      for (int i = 0; i < 8; ++i) {
        int row = mW + i2 * 16 + h16 * 8 + i;  // C layout: rows i / i+8 by lane half
        int col = nBase + j * 16 + ln;
        float v = acc[i2][j][i];
        if (bias) v += bias[col];
        if (relu) v = fmaxf(v, 0.f);
        if (res)  v += res[(size_t)row * N + col];
        if (outF) outF[(size_t)row * N + col] = v;
        else      outB[(size_t)row * N + col] = (__bf16)v;
      }
    }
  }
}

// ---------------------------------------------------------------------------
// Causal flash attention. One wave handles 16 query rows of one (b, head).
// q,k: [B,T,H,HS] bf16 ; vt: [B,H,HS,T] bf16 ; out: [B,T,H,HS] bf16.
// ---------------------------------------------------------------------------
__global__ __launch_bounds__(256) void attn_kernel(
    const __bf16* __restrict__ q, const __bf16* __restrict__ k,
    const __bf16* __restrict__ vt, __bf16* __restrict__ out) {
  const int lane = threadIdx.x & 31;
  const int wid  = threadIdx.x >> 5;
  const int h16  = lane >> 4;
  const int ln   = lane & 15;
  const int qt   = blockIdx.x * 8 + wid;
  const int bh   = blockIdx.y;
  const int b    = bh / HH;
  const int head = bh % HH;
  const int qbase = qt * 16;

  __shared__ __bf16 pbuf[8][16][32];  // per-wave P staging (C-layout -> A-layout)

  const __bf16* qr = q + (size_t)(b * TT + qbase + ln) * DD + head * HSZ;
  v16bf aq0 = pack_a(qr, h16);        // hs 0..31
  v16bf aq1 = pack_a(qr + 32, h16);   // hs 32..63

  v8f o0 = {}, o1 = {}, o2 = {}, o3 = {};
  float mrun[8], lrun[8];
#pragma unroll
  for (int i = 0; i < 8; ++i) { mrun[i] = -__builtin_inff(); lrun[i] = 0.f; }

  const int nkt = (qbase + 15) / 32 + 1;
  for (int kt = 0; kt < nkt; ++kt) {
    const int kb = kt * 32;
    const __bf16* kr0 = k + (size_t)(b * TT + kb + ln) * DD + head * HSZ;
    const __bf16* kr1 = kr0 + (size_t)16 * DD;
    v16bf bk00 = *(const v16bf*)(kr0 + h16 * 16);
    v16bf bk01 = *(const v16bf*)(kr0 + 32 + h16 * 16);
    v16bf bk10 = *(const v16bf*)(kr1 + h16 * 16);
    v16bf bk11 = *(const v16bf*)(kr1 + 32 + h16 * 16);

    v8f s0 = {}, s1 = {};
    s0 = wmma_bf16(aq0, bk00, s0);
    s0 = wmma_bf16(aq1, bk01, s0);
    s1 = wmma_bf16(aq0, bk10, s1);
    s1 = wmma_bf16(aq1, bk11, s1);

    const int col0 = kb + ln, col1 = kb + 16 + ln;
    float f0[8], f1[8], t8[8];
#pragma unroll
    for (int i = 0; i < 8; ++i) {
      int row = qbase + h16 * 8 + i;
      f0[i] = (col0 <= row) ? s0[i] * 0.125f : -__builtin_inff();
      f1[i] = (col1 <= row) ? s1[i] * 0.125f : -__builtin_inff();
      t8[i] = fmaxf(f0[i], f1[i]);
    }
#pragma unroll
    for (int i = 0; i < 8; ++i) t8[i] = grp16_max(t8[i]);

    float sc[8], rsum[8];
#pragma unroll
    for (int i = 0; i < 8; ++i) {
      float mnew = fmaxf(mrun[i], t8[i]);
      sc[i] = __expf(mrun[i] - mnew);
      mrun[i] = mnew;
      float p0 = __expf(f0[i] - mnew);
      float p1 = __expf(f1[i] - mnew);
      rsum[i] = p0 + p1;
      pbuf[wid][h16 * 8 + i][ln]      = (__bf16)p0;
      pbuf[wid][h16 * 8 + i][ln + 16] = (__bf16)p1;
    }
#pragma unroll
    for (int i = 0; i < 8; ++i) {
      rsum[i] = grp16_sum(rsum[i]);
      lrun[i] = lrun[i] * sc[i] + rsum[i];
    }
#pragma unroll
    for (int i = 0; i < 8; ++i) {
      o0[i] *= sc[i]; o1[i] *= sc[i]; o2[i] *= sc[i]; o3[i] *= sc[i];
    }

    // Wave-local LDS RAW: wait on the CDNA5 DS counter before re-reading P.
    asm volatile("s_wait_dscnt 0" ::: "memory");
    v16bf ap = pack_a(&pbuf[wid][ln][0], h16);

    const __bf16* vbase = vt + ((size_t)bh * HSZ) * TT + kb + h16 * 16;
    o0 = wmma_bf16(ap, *(const v16bf*)(vbase + (size_t)(ln) * TT),       o0);
    o1 = wmma_bf16(ap, *(const v16bf*)(vbase + (size_t)(16 + ln) * TT),  o1);
    o2 = wmma_bf16(ap, *(const v16bf*)(vbase + (size_t)(32 + ln) * TT),  o2);
    o3 = wmma_bf16(ap, *(const v16bf*)(vbase + (size_t)(48 + ln) * TT),  o3);
  }

  float inv[8];
#pragma unroll
  for (int i = 0; i < 8; ++i) inv[i] = 1.f / lrun[i];
#pragma unroll
  for (int i = 0; i < 8; ++i) {
    int row = qbase + h16 * 8 + i;
    __bf16* orow = out + (size_t)(b * TT + row) * DD + head * HSZ;
    orow[ln]      = (__bf16)(o0[i] * inv[i]);
    orow[16 + ln] = (__bf16)(o1[i] * inv[i]);
    orow[32 + ln] = (__bf16)(o2[i] * inv[i]);
    orow[48 + ln] = (__bf16)(o3[i] * inv[i]);
  }
}

// ---------------------------------------------------------------------------
// Host-side orchestration
// ---------------------------------------------------------------------------
extern "C" void kernel_launch(void* const* d_in, const int* in_sizes, int n_in,
                              void* d_out, int out_size, void* d_ws, size_t ws_size,
                              hipStream_t stream) {
  (void)in_sizes; (void)n_in; (void)out_size; (void)ws_size;
  const float* x   = (const float*)d_in[0];
  const float* Wq  = (const float*)d_in[1];
  const float* Wk  = (const float*)d_in[2];
  const float* Wv  = (const float*)d_in[3];
  const float* Wo  = (const float*)d_in[4];
  const float* bo  = (const float*)d_in[5];
  const float* W1  = (const float*)d_in[6];
  const float* b1  = (const float*)d_in[7];
  const float* W2  = (const float*)d_in[8];
  const float* b2  = (const float*)d_in[9];
  const float* g1  = (const float*)d_in[10];
  const float* be1 = (const float*)d_in[11];
  const float* g2  = (const float*)d_in[12];
  const float* be2 = (const float*)d_in[13];

  char* ws = (char*)d_ws;
  size_t off = 0;
  auto take = [&](size_t bytes) -> char* {
    char* p = ws + off;
    off += (bytes + 255) & ~(size_t)255;
    return p;
  };

  __bf16* WqT = (__bf16*)take((size_t)DD * DD * 2);
  __bf16* WkT = (__bf16*)take((size_t)DD * DD * 2);
  __bf16* WvT = (__bf16*)take((size_t)DD * DD * 2);
  __bf16* WoT = (__bf16*)take((size_t)DD * DD * 2);
  __bf16* W1T = (__bf16*)take((size_t)DD * DFF * 2);
  __bf16* W2T = (__bf16*)take((size_t)DFF * DD * 2);
  __bf16* xn  = (__bf16*)take((size_t)MM * DD * 2);  // ln1 out, reused as ln2 out
  __bf16* hn  = xn;
  char*   regionA = take((size_t)4 * MM * DD * 2);   // q,k,v,vt -> later act
  __bf16* qb  = (__bf16*)(regionA);
  __bf16* kb  = (__bf16*)(regionA + (size_t)MM * DD * 2);
  __bf16* vb  = (__bf16*)(regionA + (size_t)2 * MM * DD * 2);
  __bf16* vtB = (__bf16*)(regionA + (size_t)3 * MM * DD * 2);
  __bf16* act = (__bf16*)(regionA);
  __bf16* att = (__bf16*)take((size_t)MM * DD * 2);
  float*  x1  = (float*)take((size_t)MM * DD * 4);

  dim3 blk(256);

  // 1) weights -> bf16 [N,K]
  {
    int tot = DD * DD;
    dim3 g((tot + 255) / 256);
    convT_kernel<<<g, blk, 0, stream>>>(Wq, WqT, DD, DD);
    convT_kernel<<<g, blk, 0, stream>>>(Wk, WkT, DD, DD);
    convT_kernel<<<g, blk, 0, stream>>>(Wv, WvT, DD, DD);
    convT_kernel<<<g, blk, 0, stream>>>(Wo, WoT, DD, DD);
    int tot2 = DD * DFF;
    dim3 g2d((tot2 + 255) / 256);
    convT_kernel<<<g2d, blk, 0, stream>>>(W1, W1T, DD, DFF);
    convT_kernel<<<g2d, blk, 0, stream>>>(W2, W2T, DFF, DD);
  }

  // 2) LN1
  ln_kernel<<<dim3(MM), blk, 0, stream>>>(x, g1, be1, xn);

  // 3) QKV projections
  dim3 gD(DD / 256, MM / 64);
  gemm_bf16_kernel<<<gD, blk, 0, stream>>>(xn, WqT, nullptr, nullptr, nullptr, qb, MM, DD, DD, 0);
  gemm_bf16_kernel<<<gD, blk, 0, stream>>>(xn, WkT, nullptr, nullptr, nullptr, kb, MM, DD, DD, 0);
  gemm_bf16_kernel<<<gD, blk, 0, stream>>>(xn, WvT, nullptr, nullptr, nullptr, vb, MM, DD, DD, 0);

  // 4) V -> [B,H,HS,T]
  transV_kernel<<<dim3((MM * DD + 255) / 256), blk, 0, stream>>>(vb, vtB);

  // 5) causal flash attention
  attn_kernel<<<dim3(TT / 16 / 8, BB * HH), blk, 0, stream>>>(qb, kb, vtB, att);

  // 6) output projection + bias + residual -> x1 (f32)
  gemm_bf16_kernel<<<gD, blk, 0, stream>>>(att, WoT, bo, x, x1, nullptr, MM, DD, DD, 0);

  // 7) LN2
  ln_kernel<<<dim3(MM), blk, 0, stream>>>(x1, g2, be2, hn);

  // 8) FFN1: relu(hn @ W1 + b1) -> act (bf16)
  dim3 gF(DFF / 256, MM / 64);
  gemm_bf16_kernel<<<gF, blk, 0, stream>>>(hn, W1T, b1, nullptr, nullptr, act, MM, DFF, DD, 1);

  // 9) FFN2 + residual -> d_out (f32)
  gemm_bf16_kernel<<<gD, blk, 0, stream>>>(act, W2T, b2, x1, (float*)d_out, nullptr, MM, DD, DFF, 0);
}
